// LocalTokenTransformer_60017872994829
// MI455X (gfx1250) — matrix operand order repaired and checked by reading, not compiled, not run on hardware
//
#include <hip/hip_runtime.h>
#include <hip/hip_bf16.h>

// ---------------- problem constants ----------------
#define NTOK 1024
#define CT   768
#define CS   384
#define CP   16
#define HH   16
#define DH   48
#define NB   4
#define NL   8
#define NK   32
#define KK   40
#define TWO_CT 1536
#define SCALE_ATT 0.14433756729740643f  // 1/sqrt(48)

// ---------------- WMMA types ----------------
typedef __attribute__((ext_vector_type(16))) __bf16 bf16x16;
typedef __attribute__((ext_vector_type(8)))  float  f32x8;

union BfFrag { bf16x16 v; unsigned short u[16]; };

__device__ __forceinline__ unsigned short f2bf(float x) {
  union { float f; unsigned int u; } c; c.f = x;
  unsigned int u = c.u;
  unsigned int r = u + 0x7FFFu + ((u >> 16) & 1u);   // round to nearest even
  return (unsigned short)(r >> 16);
}

__device__ __forceinline__ float sigmoidf_(float x) {
  return 1.0f / (1.0f + __expf(-x));
}

// ---------------- generic bf16 WMMA GEMM: C = A[MxK] * W[KxN] (+bias) -------
// Tile 128x64, BK=64 (two WMMA K-substeps per LDS stage), 256 threads = 8
// waves. Wave w owns rows [w*16, w*16+16) of the tile and 4 accumulators
// (16x16 each) along N -> 8 v_wmma issues per barrier interval.
#define BM 128
#define BN 64
#define BKT 64
#define LDA_S 72   // ushort stride (64 + 8): rows land on distinct banks
#define LDB_S 72

__global__ __launch_bounds__(256)
void gemm_bf16_wmma(const float* __restrict__ A, const float* __restrict__ W,
                    const float* __restrict__ bias, float* __restrict__ C,
                    int Kd, int Nc) {
  __shared__ unsigned short As[BM * LDA_S];   // 18.4 KB
  __shared__ unsigned short Bs[BKT * LDB_S];  //  9.2 KB

  const int tid  = threadIdx.x;
  const int wave = tid >> 5;
  const int lane = tid & 31;
  const int m0   = blockIdx.x * BM;
  const int n0   = blockIdx.y * BN;

  f32x8 acc[4] = {};

  const int mrow = wave * 16 + (lane & 15);  // local A row for this lane
  const int g    = lane >> 4;                // lane group 0/1

  for (int k0 = 0; k0 < Kd; k0 += BKT) {
    // stage A tile (128x64) f32 -> bf16, float4 global loads
    #pragma unroll
    for (int r = 0; r < (BM * BKT) / (256 * 4); ++r) {   // 8
      int e = tid + r * 256;
      int row = e >> 4, c4 = e & 15;
      const float4 v = *(const float4*)(A + (size_t)(m0 + row) * Kd + k0 + c4 * 4);
      unsigned int u01 = (unsigned int)f2bf(v.x) | ((unsigned int)f2bf(v.y) << 16);
      unsigned int u23 = (unsigned int)f2bf(v.z) | ((unsigned int)f2bf(v.w) << 16);
      unsigned int* p = (unsigned int*)&As[row * LDA_S + c4 * 4];
      p[0] = u01; p[1] = u23;
    }
    // stage B tile (64x64) f32 -> bf16
    #pragma unroll
    for (int r = 0; r < (BKT * BN) / (256 * 4); ++r) {   // 4
      int e = tid + r * 256;
      int row = e >> 4, c4 = e & 15;
      const float4 v = *(const float4*)(W + (size_t)(k0 + row) * Nc + n0 + c4 * 4);
      unsigned int u01 = (unsigned int)f2bf(v.x) | ((unsigned int)f2bf(v.y) << 16);
      unsigned int u23 = (unsigned int)f2bf(v.z) | ((unsigned int)f2bf(v.w) << 16);
      unsigned int* p = (unsigned int*)&Bs[row * LDB_S + c4 * 4];
      p[0] = u01; p[1] = u23;
    }

    // prefetch next K tiles into cache while this stage computes
    if (k0 + BKT < Kd) {
      {
        int row = tid >> 1, half = tid & 1;   // 128 rows x 2 cachelines (256B/row)
        __builtin_prefetch(A + (size_t)(m0 + row) * Kd + (k0 + BKT) + half * 32, 0, 1);
      }
      if (tid < 128) {
        int row = tid >> 1, half = tid & 1;   // 64 rows x 2 cachelines
        __builtin_prefetch(W + (size_t)(k0 + BKT + row) * Nc + n0 + half * 32, 0, 1);
      }
    }
    __syncthreads();

    #pragma unroll
    for (int s = 0; s < 2; ++s) {
      // A fragment: ISA 16-bit A 16x32 layout, K-substep s
      BfFrag af;
      #pragma unroll
      for (int e2 = 0; e2 < 16; ++e2) {
        int j = e2 >> 1, pair = e2 & 1;
        int kk = s * 32 + ((j >> 2) << 4) + ((j & 3) << 1) + pair + g * 8;
        af.u[e2] = As[mrow * LDA_S + kk];
      }
      #pragma unroll
      for (int nt = 0; nt < 4; ++nt) {
        // B fragment: lane%16 = N column, element = K row (table order, same as A)
        BfFrag bf;
        #pragma unroll
        for (int e2 = 0; e2 < 16; ++e2)
          bf.u[e2] = Bs[(s * 32 + ((e2 >> 3) << 4) + (e2 & 7) + g * 8) * LDB_S + nt * 16 + (lane & 15)];
        acc[nt] = __builtin_amdgcn_wmma_f32_16x16x32_bf16(
            false, af.v, false, bf.v, (short)0, acc[nt], false, false);
      }
    }
    __syncthreads();
  }

  // epilogue: C/D layout — lane = N col, VGPR r -> M = r + 8*(lane/16)
  const int nloc = lane & 15;
  const int gg   = lane >> 4;
  #pragma unroll
  for (int nt = 0; nt < 4; ++nt) {
    int gn = n0 + nt * 16 + nloc;
    float bv = bias ? bias[gn] : 0.0f;
    #pragma unroll
    for (int r = 0; r < 8; ++r) {
      int gm = m0 + wave * 16 + gg * 8 + r;
      C[(size_t)gm * Nc + gn] = acc[nt][r] + bv;
    }
  }
}

// ---------------- rowwise LayerNorm (optional gain/bias) ----------------
__global__ __launch_bounds__(256)
void rowwise_ln(const float* __restrict__ x, float* __restrict__ y, int C,
                const float* __restrict__ gain, const float* __restrict__ bias) {
  __shared__ float r1[256];
  __shared__ float r2[256];
  const int row = blockIdx.x;
  const float* xr = x + (size_t)row * C;
  float s = 0.f, s2 = 0.f;
  for (int c = threadIdx.x; c < C; c += 256) { float v = xr[c]; s += v; s2 += v * v; }
  r1[threadIdx.x] = s; r2[threadIdx.x] = s2;
  __syncthreads();
  for (int st = 128; st > 0; st >>= 1) {
    if (threadIdx.x < st) { r1[threadIdx.x] += r1[threadIdx.x + st];
                            r2[threadIdx.x] += r2[threadIdx.x + st]; }
    __syncthreads();
  }
  float mean = r1[0] / (float)C;
  float var  = r2[0] / (float)C - mean * mean;
  float inv  = rsqrtf(var + 1e-5f);
  for (int c = threadIdx.x; c < C; c += 256) {
    float v = (xr[c] - mean) * inv;
    if (gain) v *= gain[c];
    if (bias) v += bias[c];
    y[(size_t)row * C + c] = v;
  }
}

// ---------------- AdaLN combine: an = sigmoid(g1) * ln_a + g2 -------------
__global__ __launch_bounds__(256)
void adaln_combine(const float* __restrict__ g1, const float* __restrict__ lna,
                   const float* __restrict__ g2, float* __restrict__ an, int total) {
  int i = blockIdx.x * 256 + threadIdx.x;
  if (i < total) an[i] = sigmoidf_(g1[i]) * lna[i] + g2[i];
}

// ---------------- SwiGLU: h1 <- silu(h1) * h2 ----------------
__global__ __launch_bounds__(256)
void swiglu_kernel(float* __restrict__ h1, const float* __restrict__ h2, int total) {
  int i = blockIdx.x * 256 + threadIdx.x;
  if (i < total) { float a = h1[i]; h1[i] = a * sigmoidf_(a) * h2[i]; }
}

// ---------------- gated residual: A += sigmoid(p) * x ----------------
__global__ __launch_bounds__(256)
void gated_residual(float* __restrict__ Aacc, const float* __restrict__ pre,
                    const float* __restrict__ x, int total) {
  int i = blockIdx.x * 256 + threadIdx.x;
  if (i < total) Aacc[i] += sigmoidf_(pre[i]) * x[i];
}

// ---------------- kNN + local window indices ----------------
__global__ __launch_bounds__(256)
void knn_kernel(const float* __restrict__ X, int* __restrict__ idx) {
  __shared__ unsigned long long key[NTOK];
  __shared__ unsigned long long red[256];
  const int n = blockIdx.x;
  const float x0 = X[n * 3 + 0], x1 = X[n * 3 + 1], x2 = X[n * 3 + 2];
  for (int m = threadIdx.x; m < NTOK; m += 256) {
    float dx = X[m * 3 + 0] - x0;
    float dy = X[m * 3 + 1] - x1;
    float dz = X[m * 3 + 2] - x2;
    float d2 = dx * dx + dy * dy + dz * dz;
    union { float f; unsigned int u; } c; c.f = d2;   // nonneg: bits are order-preserving
    key[m] = ((unsigned long long)c.u << 32) | (unsigned int)m;
  }
  if (threadIdx.x < NL) {
    int v = n - NL / 2 + threadIdx.x;
    v = v < 0 ? 0 : (v > NTOK - 1 ? NTOK - 1 : v);
    idx[n * KK + threadIdx.x] = v;
  }
  __syncthreads();
  for (int sel = 0; sel < NK; ++sel) {
    unsigned long long best = ~0ull;
    for (int m = threadIdx.x; m < NTOK; m += 256) {
      unsigned long long k = key[m];
      best = k < best ? k : best;
    }
    red[threadIdx.x] = best;
    __syncthreads();
    for (int st = 128; st > 0; st >>= 1) {
      if (threadIdx.x < st) {
        unsigned long long o = red[threadIdx.x + st];
        if (o < red[threadIdx.x]) red[threadIdx.x] = o;
      }
      __syncthreads();
    }
    unsigned long long w = red[0];
    if (threadIdx.x == 0) {
      int widx = (int)(w & 0xFFFFFFFFull);
      idx[n * KK + NL + sel] = widx;
      key[widx] = ~0ull;
    }
    __syncthreads();
  }
}

// ---------------- fused local attention (scores + pair bias + softmax + AV +
//                  per-channel gate sigmoid(an@Wg)) ----------------
__global__ __launch_bounds__(256)
void attn_kernel(const float* __restrict__ q, const float* __restrict__ kb,
                 const float* __restrict__ vb, const int* __restrict__ idx,
                 const float* __restrict__ Z, const float* __restrict__ pg,
                 const float* __restrict__ pb, const float* __restrict__ Wb,
                 const float* __restrict__ gl, float* __restrict__ og) {
  __shared__ float qs[CT];
  __shared__ float sc[HH * KK];
  const int n = blockIdx.x;
  const int t = threadIdx.x;
  const int h = t >> 4;       // head 0..15
  const int l = t & 15;       // lane-in-head 0..15 (each covers 3 of DH=48)

  for (int c = t; c < CT; c += 256) qs[c] = q[(size_t)n * CT + c];
  __syncthreads();

  for (int kkk = 0; kkk < KK; ++kkk) {
    const int j = idx[n * KK + kkk];
    const float* krow = kb + (size_t)j * CT + h * DH;
    float part = 0.f;
    #pragma unroll
    for (int r = 0; r < 3; ++r) {
      int d = l * 3 + r;
      part += qs[h * DH + d] * krow[d];
    }
    #pragma unroll
    for (int m = 8; m >= 1; m >>= 1) part += __shfl_xor(part, m, 32);

    // pair bias: LN over CP=16 of gathered Z row, dot with Wb column h
    const float* zrow = Z + ((size_t)n * NTOK + j) * CP;
    float zc = zrow[l];
    float s = zc, s2 = zc * zc;
    #pragma unroll
    for (int m = 8; m >= 1; m >>= 1) { s += __shfl_xor(s, m, 32); s2 += __shfl_xor(s2, m, 32); }
    float mean = s * (1.f / 16.f);
    float var  = s2 * (1.f / 16.f) - mean * mean;
    float lz   = (zc - mean) * rsqrtf(var + 1e-5f) * pg[l] + pb[l];
    float bp   = lz * Wb[l * HH + h];
    #pragma unroll
    for (int m = 8; m >= 1; m >>= 1) bp += __shfl_xor(bp, m, 32);

    if (l == 0) sc[h * KK + kkk] = part * SCALE_ATT + bp;
  }
  __syncthreads();

  if (l == 0) {   // serial softmax over K=40 per head
    float mx = -1e30f;
    for (int kkk = 0; kkk < KK; ++kkk) mx = fmaxf(mx, sc[h * KK + kkk]);
    float ss = 0.f;
    for (int kkk = 0; kkk < KK; ++kkk) {
      float e = __expf(sc[h * KK + kkk] - mx);
      sc[h * KK + kkk] = e; ss += e;
    }
    float inv = 1.f / ss;
    for (int kkk = 0; kkk < KK; ++kkk) sc[h * KK + kkk] *= inv;
  }
  __syncthreads();

  float o0 = 0.f, o1 = 0.f, o2 = 0.f;
  for (int kkk = 0; kkk < KK; ++kkk) {
    const int j = idx[n * KK + kkk];
    const float* vrow = vb + (size_t)j * CT + h * DH + l * 3;
    float w = sc[h * KK + kkk];
    o0 += w * vrow[0]; o1 += w * vrow[1]; o2 += w * vrow[2];
  }
  float ov[3] = {o0, o1, o2};
  #pragma unroll
  for (int r = 0; r < 3; ++r) {
    int c = h * DH + l * 3 + r;
    float gv = gl[(size_t)n * CT + c];
    og[(size_t)n * CT + c] = ov[r] * sigmoidf_(gv);
  }
}

// ---------------- host side ----------------
static inline size_t align_up(size_t x, size_t a) { return (x + a - 1) / a * a; }

extern "C" void kernel_launch(void* const* d_in, const int* in_sizes, int n_in,
                              void* d_out, int out_size, void* d_ws, size_t ws_size,
                              hipStream_t stream) {
  (void)in_sizes; (void)n_in; (void)out_size; (void)ws_size;
  const float* A_I   = (const float*)d_in[0];
  const float* S_I   = (const float*)d_in[1];
  const float* Z_II  = (const float*)d_in[2];
  const float* X_L   = (const float*)d_in[3];
  // d_in[4] = f (unused by reference)
  const float* aln1_sln_g = (const float*)d_in[5];
  const float* aln1_sw    = (const float*)d_in[6];
  const float* aln1_sb    = (const float*)d_in[7];
  const float* aln1_shw   = (const float*)d_in[8];
  const float* Wq    = (const float*)d_in[9];
  const float* bq    = (const float*)d_in[10];
  const float* Wk    = (const float*)d_in[11];
  const float* Wv    = (const float*)d_in[12];
  const float* plg   = (const float*)d_in[13];
  const float* plb   = (const float*)d_in[14];
  const float* Wb    = (const float*)d_in[15];
  const float* Wg    = (const float*)d_in[16];
  const float* Wo    = (const float*)d_in[17];
  const float* sg_w  = (const float*)d_in[18];
  const float* sg_b  = (const float*)d_in[19];
  const float* aln2_sln_g = (const float*)d_in[20];
  const float* aln2_sw    = (const float*)d_in[21];
  const float* aln2_sb    = (const float*)d_in[22];
  const float* aln2_shw   = (const float*)d_in[23];
  const float* t_w1  = (const float*)d_in[24];
  const float* t_w2  = (const float*)d_in[25];
  const float* t_w3  = (const float*)d_in[26];
  const float* tsg_w = (const float*)d_in[27];
  const float* tsg_b = (const float*)d_in[28];

  float* A = (float*)d_out;   // A accumulator lives in d_out

  // workspace layout
  char* base = (char*)d_ws;
  size_t off = 0;
  auto take = [&](size_t bytes) { void* p = base + off; off = align_up(off + bytes, 256); return p; };
  int*   idx  = (int*)  take((size_t)NTOK * KK * sizeof(int));
  float* sn   = (float*)take((size_t)NTOK * CS * sizeof(float));
  float* la   = (float*)take((size_t)NTOK * CT * sizeof(float));
  float* g1   = (float*)take((size_t)NTOK * CT * sizeof(float));
  float* g2   = (float*)take((size_t)NTOK * CT * sizeof(float));
  float* an   = (float*)take((size_t)NTOK * CT * sizeof(float));
  float* qb   = (float*)take((size_t)NTOK * CT * sizeof(float));
  float* kb   = (float*)take((size_t)NTOK * CT * sizeof(float));
  float* vb   = (float*)take((size_t)NTOK * CT * sizeof(float));
  float* glb  = (float*)take((size_t)NTOK * CT * sizeof(float));
  float* sgb  = (float*)take((size_t)NTOK * CT * sizeof(float));
  float* ogb  = (float*)take((size_t)NTOK * CT * sizeof(float));
  float* h1   = (float*)take((size_t)NTOK * TWO_CT * sizeof(float));
  float* h2   = (float*)take((size_t)NTOK * TWO_CT * sizeof(float));

  const int totCT  = NTOK * CT;
  const int tot2CT = NTOK * TWO_CT;
  const int ewCT   = (totCT + 255) / 256;
  const int ew2CT  = (tot2CT + 255) / 256;

  auto gemm = [&](const float* Ain, const float* Win, const float* bias,
                  float* Cout, int Kd, int Nc) {
    dim3 grid(NTOK / BM, Nc / BN);
    gemm_bf16_wmma<<<grid, 256, 0, stream>>>(Ain, Win, bias, Cout, Kd, Nc);
  };

  // A <- A_I
  hipMemcpyAsync(A, A_I, (size_t)totCT * sizeof(float),
                 hipMemcpyDeviceToDevice, stream);

  // neighbour indices (computed once, reused per block)
  knn_kernel<<<NTOK, 256, 0, stream>>>(X_L, idx);

  for (int i = 0; i < NB; ++i) {
    // ===== LocalAttentionPairBias =====
    rowwise_ln<<<NTOK, 256, 0, stream>>>(S_I, sn, CS, aln1_sln_g + (size_t)i * CS, nullptr);
    rowwise_ln<<<NTOK, 256, 0, stream>>>(A, la, CT, nullptr, nullptr);
    gemm(sn, aln1_sw  + (size_t)i * CS * CT, aln1_sb + (size_t)i * CT, g1, CS, CT);
    gemm(sn, aln1_shw + (size_t)i * CS * CT, nullptr,                  g2, CS, CT);
    adaln_combine<<<ewCT, 256, 0, stream>>>(g1, la, g2, an, totCT);

    gemm(an, Wq + (size_t)i * CT * CT, bq + (size_t)i * CT, qb, CT, CT);
    gemm(an, Wk + (size_t)i * CT * CT, nullptr, kb, CT, CT);
    gemm(an, Wv + (size_t)i * CT * CT, nullptr, vb, CT, CT);
    gemm(an, Wg + (size_t)i * CT * CT, nullptr, glb, CT, CT);
    gemm(S_I, sg_w + (size_t)i * CS * CT, sg_b + (size_t)i * CT, sgb, CS, CT);

    attn_kernel<<<NTOK, 256, 0, stream>>>(qb, kb, vb, idx, Z_II,
                                          plg + (size_t)i * CP, plb + (size_t)i * CP,
                                          Wb + (size_t)i * CP * HH, glb, ogb);

    gemm(ogb, Wo + (size_t)i * CT * CT, nullptr, qb /*reuse as o2*/, CT, CT);
    gated_residual<<<ewCT, 256, 0, stream>>>(A, sgb, qb, totCT);

    // ===== ConditionedTransitionBlock (SwiGLU) =====
    rowwise_ln<<<NTOK, 256, 0, stream>>>(S_I, sn, CS, aln2_sln_g + (size_t)i * CS, nullptr);
    rowwise_ln<<<NTOK, 256, 0, stream>>>(A, la, CT, nullptr, nullptr);
    gemm(sn, aln2_sw  + (size_t)i * CS * CT, aln2_sb + (size_t)i * CT, g1, CS, CT);
    gemm(sn, aln2_shw + (size_t)i * CS * CT, nullptr,                  g2, CS, CT);
    adaln_combine<<<ewCT, 256, 0, stream>>>(g1, la, g2, an, totCT);

    gemm(an, t_w1 + (size_t)i * CT * TWO_CT, nullptr, h1, CT, TWO_CT);
    gemm(an, t_w2 + (size_t)i * CT * TWO_CT, nullptr, h2, CT, TWO_CT);
    swiglu_kernel<<<ew2CT, 256, 0, stream>>>(h1, h2, tot2CT);
    gemm(h1, t_w3 + (size_t)i * TWO_CT * CT, nullptr, la /*reuse as h3*/, TWO_CT, CT);
    gemm(S_I, tsg_w + (size_t)i * CS * CT, tsg_b + (size_t)i * CT, sgb, CS, CT);
    gated_residual<<<ewCT, 256, 0, stream>>>(A, sgb, la, totCT);
  }
}
